// PretrainVideoMamba_78116865180426
// MI455X (gfx1250) — compile-verified, matrix operand order checked by reading, not run on hardware
//
#include <hip/hip_runtime.h>
#include <math.h>

#define BB 4
#define TT 8
#define NPATCH 196
#define DM 384
#define DEPTH 12
#define DIN 768
#define DSTATE 16
#define DTRANK 24
#define NVIS 320
#define ROWS (BB*NVIS)     /* 1280 */
#define ODIM 768
#define EPSV 1e-5f

typedef __attribute__((ext_vector_type(16))) __bf16 bf16x16;
typedef __attribute__((ext_vector_type(8)))  __bf16 bf16x8;
typedef __attribute__((ext_vector_type(8)))  float  f32x8;
typedef __attribute__((ext_vector_type(4)))  float  f32x4;

// ---------------------------------------------------------------- helpers
__device__ __forceinline__ float siluf(float x) { return x / (1.f + __expf(-x)); }

__device__ __forceinline__ float lane_bcast(float v, int l) {
  union { float f; int i; } a; a.f = v;
  union { int i; float f; } b;
  b.i = __builtin_amdgcn_readlane(a.i, l);
  return b.f;
}

// bf16 fragment: elements 0-7 = bf[k0+half*8 .. +7], 8-15 = bf[k0+16+half*8 .. +7]
// (ISA 7.12.2 16-bit A/B 16x32 layout). Two 16B loads, no guards.
__device__ __forceinline__ bf16x16 load_frag_bf(const __bf16* __restrict__ base,
                                                int k0, int half) {
  bf16x8 lo = *(const bf16x8*)(base + k0 + half * 8);
  bf16x8 hi = *(const bf16x8*)(base + k0 + 16 + half * 8);
  return __builtin_shufflevector(lo, hi, 0,1,2,3,4,5,6,7,8,9,10,11,12,13,14,15);
}

#define WMMA_BF16(A_, B_, C_) \
  __builtin_amdgcn_wmma_f32_16x16x32_bf16(false, (A_), false, (B_), (short)0, (C_), false, false)

// one k-slab worth of operand fragments (2 A rows x 4 B cols)
struct Frags { bf16x16 a0, a1, b0, b1, b2, b3; };

__device__ __forceinline__ void load_slab(Frags& f,
    const __bf16* __restrict__ Arow0, const __bf16* __restrict__ Arow1,
    const __bf16* __restrict__ B0, const __bf16* __restrict__ B1,
    const __bf16* __restrict__ B2, const __bf16* __restrict__ B3,
    int k0, int half) {
  f.a0 = load_frag_bf(Arow0, k0, half);
  f.a1 = load_frag_bf(Arow1, k0, half);
  f.b0 = load_frag_bf(B0, k0, half);
  f.b1 = load_frag_bf(B1, k0, half);
  f.b2 = load_frag_bf(B2, k0, half);
  f.b3 = load_frag_bf(B3, k0, half);
}

__device__ __forceinline__ void mma_slab(const Frags& f, f32x8 acc0[4], f32x8 acc1[4]) {
  acc0[0] = WMMA_BF16(f.a0, f.b0, acc0[0]);
  acc1[0] = WMMA_BF16(f.a1, f.b0, acc1[0]);
  acc0[1] = WMMA_BF16(f.a0, f.b1, acc0[1]);
  acc1[1] = WMMA_BF16(f.a1, f.b1, acc1[1]);
  acc0[2] = WMMA_BF16(f.a0, f.b2, acc0[2]);
  acc1[2] = WMMA_BF16(f.a1, f.b2, acc1[2]);
  acc0[3] = WMMA_BF16(f.a0, f.b3, acc0[3]);
  acc1[3] = WMMA_BF16(f.a1, f.b3, acc1[3]);
}

// ---------------------------------------------------------------- f32 -> bf16 convert
__global__ void cvt_bf16(const float* __restrict__ in, __bf16* __restrict__ out, long n) {
  long i = ((long)blockIdx.x * blockDim.x + threadIdx.x) * 4;
  if (i + 3 < n) {
    f32x4 v = *(const f32x4*)(in + i);
#pragma unroll
    for (int j = 0; j < 4; ++j) out[i + j] = (__bf16)v[j];
  } else {
    for (long k = i; k < n; ++k) out[k] = (__bf16)in[k];
  }
}

// pack dt_w (DEPTH x DIN x 24) -> (DEPTH x DIN x 32) bf16, zero padded
__global__ void pad_dtw_bf16(const float* __restrict__ in, __bf16* __restrict__ out) {
  long i = (long)blockIdx.x * blockDim.x + threadIdx.x;   // over DEPTH*DIN*32
  if (i >= (long)DEPTH * DIN * 32) return;
  long row = i >> 5; int c = (int)(i & 31);
  out[i] = (c < DTRANK) ? (__bf16)in[row * DTRANK + c] : (__bf16)0.f;
}

// ---------------------------------------------------------------- generic WMMA GEMM
// C[M,N] = A[M,K(lda)] * Bw[N,K(ldb)]^T (+bias)(+softplus). bf16 in, f32 acc.
// One wave: 32x64 tile = 8 WMMAs per k-slab sharing 2 A + 4 B fragments.
// Ping-pong double-buffered K loop (manually unrolled x2, no register rotation):
// next slab's b128 loads are in flight while the current slab's WMMAs execute.
// No load guards: M % 32 == 0 and K % 32 == 0 required; B rows clamped to N-1
// (garbage columns computed but never stored).
__global__ void __launch_bounds__(128) gemm_wmma(
    const __bf16* __restrict__ A, int lda,
    const __bf16* __restrict__ Bw, int ldb,
    const float* __restrict__ bias,
    float* __restrict__ C, __bf16* __restrict__ Cbf, int ldc,
    int M, int N, int K, int act)
{
  int wave = threadIdx.x >> 5, lane = threadIdx.x & 31;
  int tilesN4 = (N + 63) >> 6;
  int tilesM  = M >> 5;
  int tile = blockIdx.x * 4 + wave;
  if (tile >= tilesM * tilesN4) return;
  int tm = tile / tilesN4, tn = tile % tilesN4;
  int half = lane >> 4, lr = lane & 15;
  const __bf16* Arow0 = A + (long)(tm * 32 + lr) * lda;
  const __bf16* Arow1 = Arow0 + (long)16 * lda;
  const __bf16* Brow[4];
#pragma unroll
  for (int q = 0; q < 4; ++q) {
    int bn = tn * 64 + q * 16 + lr;
    if (bn > N - 1) bn = N - 1;                 // clamp, never predicate
    Brow[q] = Bw + (long)bn * ldb;
  }

  f32x8 acc0[4] = {}, acc1[4] = {};
  Frags F0, F1;
  load_slab(F0, Arow0, Arow1, Brow[0], Brow[1], Brow[2], Brow[3], 0, half);
  int k0 = 0;
  for (;;) {
    int k1 = k0 + 32;
    if (k1 >= K) { mma_slab(F0, acc0, acc1); break; }
    load_slab(F1, Arow0, Arow1, Brow[0], Brow[1], Brow[2], Brow[3], k1, half);
    mma_slab(F0, acc0, acc1);
    int k2 = k1 + 32;
    if (k2 >= K) { mma_slab(F1, acc0, acc1); break; }
    load_slab(F0, Arow0, Arow1, Brow[0], Brow[1], Brow[2], Brow[3], k2, half);
    mma_slab(F1, acc0, acc1);
    k0 = k2;
  }

#pragma unroll
  for (int q = 0; q < 4; ++q) {
    int n = tn * 64 + q * 16 + lr;
    if (n >= N) continue;
    float bv = bias ? bias[n] : 0.f;
#pragma unroll
    for (int mt = 0; mt < 2; ++mt) {
#pragma unroll
      for (int v = 0; v < 8; ++v) {
        int m = tm * 32 + mt * 16 + half * 8 + v;
        float x = (mt ? acc1[q][v] : acc0[q][v]) + bv;
        if (act == 1) x = (x > 20.f) ? x : log1pf(__expf(x));
        if (C)   C[(long)m * ldc + n] = x;
        if (Cbf) Cbf[(long)m * ldc + n] = (__bf16)x;
      }
    }
  }
}

// ---------------------------------------------------------------- patch-embed GEMM
// Implicit im2col + vis_idx gather; A streamed from f32 video (converted inline),
// B = pre-converted bf16 patch weights. cls rows (l==0) read a dummy patch and
// are overwritten by assemble_tokens.
__global__ void __launch_bounds__(128) patch_gemm_wmma(
    const float* __restrict__ video, const int* __restrict__ vis,
    const __bf16* __restrict__ pw, float* __restrict__ C)
{
  int wave = threadIdx.x >> 5, lane = threadIdx.x & 31;
  const int tilesN4 = DM / 64;                  // 6
  int tile = blockIdx.x * 4 + wave;
  if (tile >= (ROWS / 16) * tilesN4) return;
  int tm = tile / tilesN4, tn = tile % tilesN4;
  int half = lane >> 4, lr = lane & 15;
  int r = tm * 16 + lr;
  int b = r / NVIS;
  int l = vis[r];
  int t = 0, pixbase = 0;
  if (l > 0) {
    int li = l - 1; t = li / NPATCH; int p = li % NPATCH;
    pixbase = (p / 14) * 16 * 224 + (p % 14) * 16;
  }
  const __bf16* Brow[4];
#pragma unroll
  for (int q = 0; q < 4; ++q) Brow[q] = pw + (long)(tn * 64 + q * 16 + lr) * 768;

  f32x8 acc[4] = {};
  for (int k0 = 0; k0 < 768; k0 += 32) {
    bf16x16 af;
#pragma unroll
    for (int seg = 0; seg < 2; ++seg) {
      int ks = k0 + half * 8 + seg * 16;
      int c = ks >> 8, rem = ks & 255, py = rem >> 4, px = rem & 15;
      const float* sp = video + (long)(b * 24 + c * 8 + t) * 50176
                              + pixbase + py * 224 + px;
      f32x4 q0 = *(const f32x4*)sp;
      f32x4 q1 = *(const f32x4*)(sp + 4);
#pragma unroll
      for (int e = 0; e < 4; ++e) {
        af[seg * 8 + e]     = (__bf16)q0[e];
        af[seg * 8 + 4 + e] = (__bf16)q1[e];
      }
    }
#pragma unroll
    for (int q = 0; q < 4; ++q) {
      bf16x16 bf = load_frag_bf(Brow[q], k0, half);
      acc[q] = WMMA_BF16(af, bf, acc[q]);
    }
  }
#pragma unroll
  for (int q = 0; q < 4; ++q) {
    int n = tn * 64 + q * 16 + lr;
#pragma unroll
    for (int v = 0; v < 8; ++v) {
      int m = tm * 16 + half * 8 + v;
      C[(long)m * DM + n] = acc[q][v];
    }
  }
}

// ---------------------------------------------------------------- token assembly
__global__ void assemble_tokens(const float* __restrict__ patchC,
                                const int* __restrict__ vis,
                                const float* __restrict__ patch_b,
                                const float* __restrict__ cls_token,
                                const float* __restrict__ pos_embed,
                                const float* __restrict__ temporal_pos,
                                float* __restrict__ hid, float* __restrict__ res)
{
  int idx = blockIdx.x * blockDim.x + threadIdx.x;
  if (idx >= ROWS * DM) return;
  int r = idx / DM, d = idx - r * DM;
  int l = vis[r];
  float v;
  if (l == 0) {
    v = cls_token[d] + pos_embed[d];
  } else {
    int li = l - 1, t = li / NPATCH, p = li % NPATCH;
    v = patchC[idx] + patch_b[d] + pos_embed[(1 + p) * DM + d] + temporal_pos[t * DM + d];
  }
  hid[idx] = v;
  res[idx] = 0.f;
}

// ---------------------------------------------------------------- residual + rmsnorm (bf16 out)
__global__ void __launch_bounds__(128) rms_residual(float* __restrict__ res,
    const float* __restrict__ hid, const float* __restrict__ w, __bf16* __restrict__ h)
{
  int r = blockIdx.x;
  int lane = threadIdx.x & 31, wave = threadIdx.x >> 5;
  __shared__ float red[4];
  float v[3]; float ss = 0.f;
#pragma unroll
  for (int j = 0; j < 3; ++j) {
    int d = threadIdx.x + j * 128;
    float x = res[(long)r * DM + d] + hid[(long)r * DM + d];
    res[(long)r * DM + d] = x;
    v[j] = x; ss += x * x;
  }
#pragma unroll
  for (int o = 16; o > 0; o >>= 1) ss += __shfl_xor(ss, o, 32);
  if (lane == 0) red[wave] = ss;
  __syncthreads();
  float sc = rsqrtf((red[0] + red[1] + red[2] + red[3]) / (float)DM + EPSV);
#pragma unroll
  for (int j = 0; j < 3; ++j) {
    int d = threadIdx.x + j * 128;
    h[(long)r * DM + d] = (__bf16)(v[j] * sc * w[d]);
  }
}

// ---------------------------------------------------------------- causal dwconv + silu (bf16 out)
__global__ void dwconv_silu(const float* __restrict__ xz, const float* __restrict__ cw,
                            const float* __restrict__ cb, __bf16* __restrict__ xc, int dir)
{
  int idx = blockIdx.x * blockDim.x + threadIdx.x;
  if (idx >= ROWS * DIN) return;
  int r = idx / DIN, d = idx - r * DIN;
  int b = r / NVIS, i = r - b * NVIS;
  f32x4 wv = *(const f32x4*)(cw + d * 4);
  float s = cb[d];
  long base = (long)b * NVIS;
  if (i >= 3) {
#pragma unroll
    for (int j = 0; j < 4; ++j) {
      int m = i - 3 + j;
      int oi = dir ? (NVIS - 1 - m) : m;
      s = fmaf(wv[j], xz[(base + oi) * (2 * DIN) + d], s);
    }
  } else {
#pragma unroll
    for (int j = 0; j < 4; ++j) {
      int m = i - 3 + j;
      if (m >= 0) {
        int oi = dir ? (NVIS - 1 - m) : m;
        s = fmaf(wv[j], xz[(base + oi) * (2 * DIN) + d], s);
      }
    }
  }
  xc[idx] = (__bf16)siluf(s);
}

// ---------------------------------------------------------------- selective scan
// One lane per channel; 16 f32 states in VGPRs; wave-uniform B/C broadcast via
// v_readlane (SGPR-operand FMAs). dir==0 writes f32 partial; dir==1 adds and
// emits the bf16 operand for the out_proj WMMA GEMM.
__global__ void __launch_bounds__(128) sscan(const __bf16* __restrict__ xc,
    const float* __restrict__ dbl, const float* __restrict__ dtv_,
    const float* __restrict__ xz, const float* __restrict__ Al,
    const float* __restrict__ Dl, float* __restrict__ yf,
    __bf16* __restrict__ ybf, int dir)
{
  int wave = threadIdx.x >> 5, lane = threadIdx.x & 31;
  int gw = blockIdx.x * 4 + wave;
  if (gw >= BB * (DIN / 32)) return;
  int b = gw / (DIN / 32);
  int d = (gw % (DIN / 32)) * 32 + lane;

  float Aa[DSTATE], h[DSTATE];
#pragma unroll
  for (int s = 0; s < DSTATE; ++s) { Aa[s] = -__expf(Al[d * DSTATE + s]); h[s] = 0.f; }
  float Dval = Dl[d];

  for (int k = 0; k < NVIS; ++k) {
    long rk = (long)b * NVIS + k;
    float dt = dtv_[rk * DIN + d];
    float xv = (float)xc[rk * DIN + d];
    float bc = dbl[rk * 56 + 24 + lane];       // lanes 0-15: B, 16-31: C
    float acc = 0.f;
#pragma unroll
    for (int s = 0; s < DSTATE; ++s) {
      float Bs = lane_bcast(bc, s);
      float Cs = lane_bcast(bc, 16 + s);
      float dA = __expf(dt * Aa[s]);
      h[s] = fmaf(dA, h[s], dt * Bs * xv);
      acc = fmaf(h[s], Cs, acc);
    }
    float outv = acc + xv * Dval;
    int oi = dir ? (NVIS - 1 - k) : k;
    long ri = (long)(b * NVIS + oi);
    float zv = xz[ri * (2 * DIN) + DIN + d];
    float val = outv * siluf(zv);
    if (dir) ybf[ri * DIN + d] = (__bf16)(yf[ri * DIN + d] + val);
    else     yf[ri * DIN + d] = val;
  }
}

// ---------------------------------------------------------------- final layernorm
__global__ void __launch_bounds__(256) layernorm_out(const float* __restrict__ x,
    const float* __restrict__ w, const float* __restrict__ bb, float* __restrict__ out)
{
  int r = blockIdx.x;
  int lane = threadIdx.x & 31, wave = threadIdx.x >> 5;
  __shared__ float rs[8], rq[8];
  float v[3]; float s = 0.f, q = 0.f;
#pragma unroll
  for (int j = 0; j < 3; ++j) {
    int d = threadIdx.x + j * 256;
    float t = x[(long)r * ODIM + d];
    v[j] = t; s += t; q += t * t;
  }
#pragma unroll
  for (int o = 16; o > 0; o >>= 1) { s += __shfl_xor(s, o, 32); q += __shfl_xor(q, o, 32); }
  if (lane == 0) { rs[wave] = s; rq[wave] = q; }
  __syncthreads();
  float S = 0.f, Q = 0.f;
#pragma unroll
  for (int i = 0; i < 8; ++i) { S += rs[i]; Q += rq[i]; }
  float mean = S / (float)ODIM;
  float var = Q / (float)ODIM - mean * mean;
  float sc = rsqrtf(var + EPSV);
#pragma unroll
  for (int j = 0; j < 3; ++j) {
    int d = threadIdx.x + j * 256;
    out[(long)r * ODIM + d] = (v[j] - mean) * sc * w[d] + bb[d];
  }
}

// ---------------------------------------------------------------- host driver
extern "C" void kernel_launch(void* const* d_in, const int* in_sizes, int n_in,
                              void* d_out, int out_size, void* d_ws, size_t ws_size,
                              hipStream_t stream)
{
  (void)in_sizes; (void)n_in; (void)out_size; (void)ws_size;
  const float* video        = (const float*)d_in[0];
  const int*   vis          = (const int*)  d_in[1];
  const float* patch_w      = (const float*)d_in[2];
  const float* patch_b      = (const float*)d_in[3];
  const float* cls_token    = (const float*)d_in[4];
  const float* pos_embed    = (const float*)d_in[5];
  const float* temporal_pos = (const float*)d_in[6];
  const float* in_proj_w    = (const float*)d_in[7];
  const float* conv_w       = (const float*)d_in[8];
  const float* conv_b       = (const float*)d_in[9];
  const float* conv_b_w     = (const float*)d_in[10];
  const float* conv_b_b     = (const float*)d_in[11];
  const float* x_proj_w     = (const float*)d_in[12];
  const float* x_proj_b_w   = (const float*)d_in[13];
  const float* dt_w         = (const float*)d_in[14];
  const float* dt_b         = (const float*)d_in[15];
  const float* dt_b_w       = (const float*)d_in[16];
  const float* dt_b_b       = (const float*)d_in[17];
  const float* A_log        = (const float*)d_in[18];
  const float* A_b_log      = (const float*)d_in[19];
  const float* Dv           = (const float*)d_in[20];
  const float* D_b          = (const float*)d_in[21];
  const float* norm_w       = (const float*)d_in[22];
  const float* out_proj_w   = (const float*)d_in[23];
  const float* norm_f_w     = (const float*)d_in[24];
  const float* head_w       = (const float*)d_in[25];
  const float* head_b       = (const float*)d_in[26];
  const float* dec_w        = (const float*)d_in[27];
  const float* dec_b        = (const float*)d_in[28];

  // ---------------- workspace carve-up (all chunks 16B aligned) ----------------
  float* F = (float*)d_ws;
  float* hid   = F; F += (long)ROWS * DM;        // f32
  float* res   = F; F += (long)ROWS * DM;
  float* xz    = F; F += (long)ROWS * 2 * DIN;   // also final head output (ldc 768)
  float* dblf  = F; F += (long)ROWS * 56;
  float* dtbuf = F; F += (long)ROWS * DIN;       // also patch-GEMM f32 output
  float* yf    = F; F += (long)ROWS * DIN;

  __bf16* G = (__bf16*)F;
  __bf16* hbf   = G; G += (long)ROWS * DM;
  __bf16* xcbf  = G; G += (long)ROWS * DIN;
  __bf16* dblbf = G; G += (long)ROWS * 56;
  __bf16* ybf   = G; G += (long)ROWS * DIN;
  __bf16* wpat  = G; G += (long)DM * DIN;               // patch_w
  __bf16* winp  = G; G += (long)DEPTH * 2 * DIN * DM;   // in_proj
  __bf16* wxp0  = G; G += (long)DEPTH * 56 * DIN;
  __bf16* wxp1  = G; G += (long)DEPTH * 56 * DIN;
  __bf16* wdt0  = G; G += (long)DEPTH * DIN * 32;       // zero-padded K=32
  __bf16* wdt1  = G; G += (long)DEPTH * DIN * 32;
  __bf16* wout  = G; G += (long)DEPTH * DM * DIN;
  __bf16* whead = G; G += (long)ODIM * DM;

  auto cvtGrid = [](long n) { return (int)((n / 4 + 255) / 256); };
  auto gemmGrid = [](int M, int N) {
    int t = (M / 32) * ((N + 63) / 64);
    return (t + 3) / 4;
  };
  auto patchGrid = []() {
    int t = (ROWS / 16) * (DM / 64);
    return (t + 3) / 4;
  };

  // ---------------- one-time (per launch) weight conversion to bf16 ----------------
  cvt_bf16<<<cvtGrid((long)DM * DIN), 256, 0, stream>>>(patch_w, wpat, (long)DM * DIN);
  cvt_bf16<<<cvtGrid((long)DEPTH * 2 * DIN * DM), 256, 0, stream>>>(in_proj_w, winp, (long)DEPTH * 2 * DIN * DM);
  cvt_bf16<<<cvtGrid((long)DEPTH * 56 * DIN), 256, 0, stream>>>(x_proj_w, wxp0, (long)DEPTH * 56 * DIN);
  cvt_bf16<<<cvtGrid((long)DEPTH * 56 * DIN), 256, 0, stream>>>(x_proj_b_w, wxp1, (long)DEPTH * 56 * DIN);
  cvt_bf16<<<cvtGrid((long)DEPTH * DM * DIN), 256, 0, stream>>>(out_proj_w, wout, (long)DEPTH * DM * DIN);
  cvt_bf16<<<cvtGrid((long)ODIM * DM), 256, 0, stream>>>(head_w, whead, (long)ODIM * DM);
  pad_dtw_bf16<<<(int)(((long)DEPTH * DIN * 32 + 255) / 256), 256, 0, stream>>>(dt_w, wdt0);
  pad_dtw_bf16<<<(int)(((long)DEPTH * DIN * 32 + 255) / 256), 256, 0, stream>>>(dt_b_w, wdt1);

  // ---------------- patch embed (implicit im2col + gather) + token assembly ----------------
  patch_gemm_wmma<<<patchGrid(), 128, 0, stream>>>(video, vis, wpat, dtbuf);
  assemble_tokens<<<(ROWS * DM + 255) / 256, 256, 0, stream>>>(
      dtbuf, vis, patch_b, cls_token, pos_embed, temporal_pos, hid, res);

  for (int layer = 0; layer < DEPTH; ++layer) {
    rms_residual<<<ROWS, 128, 0, stream>>>(res, hid, norm_w + (long)layer * DM, hbf);
    gemm_wmma<<<gemmGrid(ROWS, 2 * DIN), 128, 0, stream>>>(
        hbf, DM, winp + (long)layer * 2 * DIN * DM, DM,
        nullptr, xz, nullptr, 2 * DIN, ROWS, 2 * DIN, DM, 0);

    for (int dir = 0; dir < 2; ++dir) {
      const float*  cw  = dir ? conv_b_w : conv_w;
      const float*  cb  = dir ? conv_b_b : conv_b;
      const __bf16* xpw = dir ? wxp1 : wxp0;
      const __bf16* dw  = dir ? wdt1 : wdt0;
      const float*  db  = dir ? dt_b_b : dt_b;
      const float*  Al  = dir ? A_b_log : A_log;
      const float*  Dl  = dir ? D_b : Dv;

      dwconv_silu<<<(ROWS * DIN + 255) / 256, 256, 0, stream>>>(
          xz, cw + (long)layer * DIN * 4, cb + (long)layer * DIN, xcbf, dir);
      gemm_wmma<<<gemmGrid(ROWS, 56), 128, 0, stream>>>(
          xcbf, DIN, xpw + (long)layer * 56 * DIN, DIN,
          nullptr, dblf, dblbf, 56, ROWS, 56, DIN, 0);
      gemm_wmma<<<gemmGrid(ROWS, DIN), 128, 0, stream>>>(
          dblbf, 56, dw + (long)layer * DIN * 32, 32,
          db + (long)layer * DIN, dtbuf, nullptr, DIN, ROWS, DIN, 32, 1);
      sscan<<<(BB * (DIN / 32) + 3) / 4, 128, 0, stream>>>(
          xcbf, dblf, dtbuf, xz, Al + (long)layer * DIN * DSTATE,
          Dl + (long)layer * DIN, yf, ybf, dir);
    }
    gemm_wmma<<<gemmGrid(ROWS, DM), 128, 0, stream>>>(
        ybf, DIN, wout + (long)layer * DM * DIN, DIN,
        nullptr, hid, nullptr, DM, ROWS, DM, DIN, 0);
  }

  // ---------------- final rmsnorm -> head GEMM (+bias) -> layernorm ----------------
  rms_residual<<<ROWS, 128, 0, stream>>>(res, hid, norm_f_w, hbf);
  gemm_wmma<<<gemmGrid(ROWS, ODIM), 128, 0, stream>>>(
      hbf, DM, whead, DM, head_b, xz, nullptr, ODIM, ROWS, ODIM, DM, 0);
  layernorm_out<<<ROWS, 256, 0, stream>>>(xz, dec_w, dec_b, (float*)d_out);
}
